// GCNLayer_6347961663936
// MI455X (gfx1250) — compile-verified, matrix operand order checked by reading, not run on hardware
//
#include <hip/hip_runtime.h>

typedef _Float16 v8h  __attribute__((ext_vector_type(8)));
typedef _Float16 v16h __attribute__((ext_vector_type(16)));
typedef float    v8f  __attribute__((ext_vector_type(8)));

#define NNODES 8192

// ---------------------------------------------------------------------------
// 1) d[i] = rsqrt(1 + sum_j adj[i][j])   (diag of adj is already zero)
// ---------------------------------------------------------------------------
__global__ void deg_rsqrt_kernel(const float* __restrict__ adj,
                                 float* __restrict__ dvec) {
  __shared__ float red[256];
  const int row = blockIdx.x;
  const float4* rp = (const float4*)(adj + (size_t)row * NNODES);
  float s = 0.f;
  for (int j = threadIdx.x; j < NNODES / 4; j += 256) {
    float4 v = rp[j];
    s += v.x + v.y + v.z + v.w;
  }
  red[threadIdx.x] = s;
  __syncthreads();
  for (int off = 128; off > 0; off >>= 1) {
    if (threadIdx.x < off) red[threadIdx.x] += red[threadIdx.x + off];
    __syncthreads();
  }
  if (threadIdx.x == 0) {
    float dg = red[0] + 1.0f;  // self-loop
    dvec[row] = dg > 0.f ? rsqrtf(dg) : 0.f;
  }
}

// ---------------------------------------------------------------------------
// 2,5) Out[m][n] = sum_c X[m][c] * W[c][n]   (one block per row m, blockDim=Nc)
// ---------------------------------------------------------------------------
__global__ void rowgemm_kernel(const float* __restrict__ X,
                               const float* __restrict__ W,
                               float* __restrict__ Out, int Kc, int Nc) {
  __shared__ float xrow[256];
  const int m = blockIdx.x;
  for (int c = threadIdx.x; c < Kc; c += blockDim.x)
    xrow[c] = X[(size_t)m * Kc + c];
  __syncthreads();
  const int n = threadIdx.x;
  float s = 0.f;
  for (int c = 0; c < Kc; ++c) s = fmaf(xrow[c], W[c * Nc + n], s);
  Out[(size_t)m * Nc + n] = s;
}

// ---------------------------------------------------------------------------
// 3,6) dst[c][r] = f16( d[r] * src[r][c] ),  src: [8192][C] f32, dst: [C][8192] f16
// ---------------------------------------------------------------------------
__global__ void transpose_scale_f16_kernel(const float* __restrict__ src,
                                           _Float16* __restrict__ dst,
                                           const float* __restrict__ dvec, int C) {
  __shared__ float tile[32][33];
  const int r0 = blockIdx.x * 32, c0 = blockIdx.y * 32;
  const int tx = threadIdx.x, ty = threadIdx.y;  // block (32,8)
  #pragma unroll
  for (int i = 0; i < 4; ++i)
    tile[ty + 8 * i][tx] = src[(size_t)(r0 + ty + 8 * i) * C + (c0 + tx)];
  __syncthreads();
  const float dr = dvec[r0 + tx];
  #pragma unroll
  for (int i = 0; i < 4; ++i) {
    int c = c0 + ty + 8 * i;
    dst[(size_t)c * NNODES + (r0 + tx)] = (_Float16)(dr * tile[tx][ty + 8 * i]);
  }
}

// ---------------------------------------------------------------------------
// 4,7) Out = D * (A @ Bt^T + Bt^T) + bias  [optionally relu]
//   A:   [8192][8192] fp32 raw adjacency (diag==0), converted to f16 on the fly
//   Bt:  [ncols][8192] f16, rows pre-scaled by d_k  (B' = D*B, transposed)
//   Each block owns a 16-row M strip; wave w owns 64 output columns (4 WMMA tiles).
//   blockDim.x must be 32 * (ncols/64).
// ---------------------------------------------------------------------------
__global__ void an_wmma_kernel(const float* __restrict__ A,
                               const _Float16* __restrict__ Bt,
                               const float* __restrict__ dvec,
                               const float* __restrict__ bias,
                               float* __restrict__ Out, int ncols, int relu) {
  const int lane  = threadIdx.x & 31;
  const int wave  = threadIdx.x >> 5;
  const int lr    = lane & 15;   // row within 16-lane group
  const int half  = lane >> 4;   // K-octet selector
  const int mBase = blockIdx.x * 16;
  const int nBase = wave * 64;

  // A fragment source: lane (l, l+16) both read row mBase+lr, different K octets
  const float* arow = A + (size_t)(mBase + lr) * NNODES;
  // B fragment source: lane lr reads row nBase + t*16 + lr of Bt
  const _Float16* brow0 = Bt + (size_t)(nBase + lr) * NNODES;

  v8f acc[4] = {};

  for (int k0 = 0; k0 < NNODES; k0 += 32) {
    const int ka = k0 + half * 8;  // this lane's first K octet

    // A fragment: K slots {ka..ka+7} and {ka+16..ka+23}, cvt fp32 -> f16
    float4 f0 = *(const float4*)(arow + ka);
    float4 f1 = *(const float4*)(arow + ka + 4);
    float4 f2 = *(const float4*)(arow + ka + 16);
    float4 f3 = *(const float4*)(arow + ka + 20);
    v16h af;
    af[0]  = (_Float16)f0.x; af[1]  = (_Float16)f0.y;
    af[2]  = (_Float16)f0.z; af[3]  = (_Float16)f0.w;
    af[4]  = (_Float16)f1.x; af[5]  = (_Float16)f1.y;
    af[6]  = (_Float16)f1.z; af[7]  = (_Float16)f1.w;
    af[8]  = (_Float16)f2.x; af[9]  = (_Float16)f2.y;
    af[10] = (_Float16)f2.z; af[11] = (_Float16)f2.w;
    af[12] = (_Float16)f3.x; af[13] = (_Float16)f3.y;
    af[14] = (_Float16)f3.z; af[15] = (_Float16)f3.w;

    #pragma unroll
    for (int t = 0; t < 4; ++t) {
      const _Float16* br = brow0 + (size_t)t * 16 * NNODES;
      v8h blo = *(const v8h*)(br + ka);        // 16B aligned
      v8h bhi = *(const v8h*)(br + ka + 16);
      v16h bf = __builtin_shufflevector(blo, bhi, 0, 1, 2, 3, 4, 5, 6, 7,
                                        8, 9, 10, 11, 12, 13, 14, 15);
      acc[t] = __builtin_amdgcn_wmma_f32_16x16x32_f16(
          /*neg_a=*/false, af, /*neg_b=*/false, bf,
          /*c_mod=*/(short)0, acc[t], /*reuse_a=*/false, /*reuse_b=*/false);
    }
  }

  // Epilogue: out[m][n] = d_m * (acc + B'[m][n]) + bias[n]  (self-loop exact, fp32)
  #pragma unroll
  for (int t = 0; t < 4; ++t) {
    const int n = nBase + t * 16 + lr;
    const _Float16* bcol = Bt + (size_t)n * NNODES;
    const float bn = bias[n];
    #pragma unroll
    for (int r = 0; r < 8; ++r) {
      const int m = mBase + half * 8 + r;
      float v = dvec[m] * (acc[t][r] + (float)bcol[m]) + bn;
      if (relu) v = fmaxf(v, 0.f);
      Out[(size_t)m * ncols + n] = v;
    }
  }
}

// ---------------------------------------------------------------------------
extern "C" void kernel_launch(void* const* d_in, const int* in_sizes, int n_in,
                              void* d_out, int out_size, void* d_ws, size_t ws_size,
                              hipStream_t stream) {
  (void)in_sizes; (void)n_in; (void)out_size; (void)ws_size;
  const float* x   = (const float*)d_in[0];  // [8192][128]
  const float* adj = (const float*)d_in[1];  // [8192][8192]
  const float* W1  = (const float*)d_in[2];  // [128][256]
  const float* b1  = (const float*)d_in[3];  // [256]
  const float* W2  = (const float*)d_in[4];  // [256][128]
  const float* b2  = (const float*)d_in[5];  // [128]
  float* out = (float*)d_out;                // [8192][128]

  char* ws = (char*)d_ws;
  float*    dvec = (float*)ws;                                       // 32 KB
  _Float16* B1t  = (_Float16*)(ws + (32 << 10));                     // 4 MB
  float*    H    = (float*)(ws + (32 << 10) + (4 << 20));            // 8 MB
  _Float16* B2t  = (_Float16*)(ws + (32 << 10) + (12 << 20));        // 2 MB
  float*    TMP  = (float*)(ws + (32 << 10) + (14 << 20));           // 8 MB

  // 1) degrees -> d = rsqrt(deg)
  deg_rsqrt_kernel<<<NNODES, 256, 0, stream>>>(adj, dvec);

  // 2) XW1 = x @ W1 (fp32)
  rowgemm_kernel<<<NNODES, 256, 0, stream>>>(x, W1, TMP, 128, 256);

  // 3) B1t[n][k] = f16(d_k * XW1[k][n])
  transpose_scale_f16_kernel<<<dim3(NNODES / 32, 256 / 32), dim3(32, 8), 0, stream>>>(
      TMP, B1t, dvec, 256);

  // 4) H = relu(D*(A@B1' + B1') + b1)   — WMMA layer 1 (4 waves/block: 256 cols)
  an_wmma_kernel<<<NNODES / 16, 128, 0, stream>>>(adj, B1t, dvec, b1, H, 256, 1);

  // 5) HW2 = H @ W2 (fp32)
  rowgemm_kernel<<<NNODES, 128, 0, stream>>>(H, W2, TMP, 256, 128);

  // 6) B2t[n][k] = f16(d_k * HW2[k][n])
  transpose_scale_f16_kernel<<<dim3(NNODES / 32, 128 / 32), dim3(32, 8), 0, stream>>>(
      TMP, B2t, dvec, 128);

  // 7) out = D*(A@B2' + B2') + b2       — WMMA layer 2 (2 waves/block: 128 cols)
  an_wmma_kernel<<<NNODES / 16, 64, 0, stream>>>(adj, B2t, dvec, b2, out, 128, 0);
}
